// SwinV23DMHSA_26371099197931
// MI455X (gfx1250) — matrix operand order repaired and checked by reading, not compiled, not run on hardware
//
#include <hip/hip_runtime.h>

// ---------------------------------------------------------------------------
// SwinV2-3D windowed MHSA for MI455X (gfx1250, wave32, WMMA bf16 16x16x32)
// Compute-bound (~26 GFLOP vs ~7us of HBM traffic at 23.3 TB/s), so all four
// matmuls run on v_wmma_f32_16x16x32_bf16 (8x the K-depth of the f32 WMMA).
// Tile staging uses GLOBAL_LOAD_ASYNC_TO_LDS_B128 (ASYNCcnt) to bypass VGPRs.
// Softmax / L2-normalize / CPB bias MLP stay in f32 VALU.
// ---------------------------------------------------------------------------

#define NWIN   128   // windows
#define NTOK   512   // tokens per window (8*8*8)
#define DIM    128
#define HEADS  4
#define HDIM   32    // DIM / HEADS
#define QTILE  64    // query rows per attention block
#define TBL    3375  // 15*15*15 cpb table entries

typedef __attribute__((ext_vector_type(16))) __bf16 v16bf;
typedef __attribute__((ext_vector_type(8)))  float  v8f;

union FragBF { v16bf v; unsigned int u[8]; };

// Async global->LDS copy, 16 bytes per lane (GVS mode: saddr + 32-bit voffset).
__device__ __forceinline__ void async_copy_b128(unsigned lds_off, const void* gbase,
                                                unsigned gbyte_off) {
  asm volatile("global_load_async_to_lds_b128 %0, %1, %2"
               :: "v"(lds_off), "v"(gbyte_off), "s"(gbase)
               : "memory");
}
__device__ __forceinline__ void wait_async0() {
  asm volatile("s_wait_asynccnt 0x0" ::: "memory");
}

// A-matrix 16x32 bf16 fragment (ISA 7.12.2 "16-bit A-Matrix 16x32").
// base points at (row0,k0) of a row-major [M][ld] bf16 array (LDS).
__device__ __forceinline__ v16bf load_a_frag(const __bf16* base, int ld, int lane) {
  int m  = lane & 15;
  int kb = (lane & 16) ? 8 : 0;
  const __bf16* p = base + m * ld;
  FragBF f;
#pragma unroll
  for (int j = 0; j < 4; ++j) {
    f.u[j]     = *(const unsigned int*)(p + kb + 2 * j);
    f.u[4 + j] = *(const unsigned int*)(p + 16 + kb + 2 * j);
  }
  return f.v;
}

// B-matrix 32x16 bf16 fragment, loaded from B^T stored row-major [N][ld]
// (K striping matches the A fragment: lanes 0-15 K=0..7,16..23; 16-31 K=8..15,24..31).
__device__ __forceinline__ v16bf load_b_frag(const __bf16* baseT, int ld, int lane) {
  int n  = lane & 15;
  int kb = (lane & 16) ? 8 : 0;
  const __bf16* p = baseT + n * ld + kb;
  FragBF f;
#pragma unroll
  for (int j = 0; j < 4; ++j) {
    f.u[j]     = *(const unsigned int*)(p + 2 * j);
    f.u[4 + j] = *(const unsigned int*)(p + 16 + 2 * j);
  }
  return f.v;
}

__device__ __forceinline__ v8f wmma_bf16(v16bf a, v16bf b, v8f c) {
  return __builtin_amdgcn_wmma_f32_16x16x32_bf16(false, a, false, b, (short)0, c,
                                                 false, false);
}

// ---------------------------------------------------------------------------
// K0: fp32 -> bf16 staging of X, Wqkv, Wproj
// ---------------------------------------------------------------------------
__global__ void prep_kernel(const float* __restrict__ hs,
                            const float* __restrict__ wqkv,
                            const float* __restrict__ wproj,
                            __bf16* __restrict__ xbf,
                            __bf16* __restrict__ wqb,
                            __bf16* __restrict__ wpb) {
  const int NX = NWIN * NTOK * DIM;      // 8388608
  const int NQ = 3 * DIM * DIM;          // 49152
  const int NP = DIM * DIM;              // 16384
  int i = blockIdx.x * blockDim.x + threadIdx.x;
  if (i < NX)                xbf[i]            = (__bf16)hs[i];
  else if (i < NX + NQ)      wqb[i - NX]       = (__bf16)wqkv[i - NX];
  else if (i < NX + NQ + NP) wpb[i - NX - NQ]  = (__bf16)wproj[i - NX - NQ];
}

// ---------------------------------------------------------------------------
// K1: continuous position bias MLP -> bias_table[3375][HEADS] (pre-sigmoid)
// ---------------------------------------------------------------------------
__global__ void cpb_kernel(const float* __restrict__ tbl,   // [3375][3]
                           const float* __restrict__ w1,    // [512][3]
                           const float* __restrict__ b1,    // [512]
                           const float* __restrict__ w2,    // [4][512]
                           float* __restrict__ bt) {        // [3375][4]
  int e = blockIdx.x;
  int t = threadIdx.x;                                      // 128 threads
  float x0 = tbl[e * 3 + 0], x1 = tbl[e * 3 + 1], x2 = tbl[e * 3 + 2];
  float p0 = 0.f, p1 = 0.f, p2 = 0.f, p3 = 0.f;
  for (int j = t; j < 512; j += 128) {
    float h = fmaf(w1[j * 3 + 2], x2, fmaf(w1[j * 3 + 1], x1, fmaf(w1[j * 3], x0, b1[j])));
    h = fmaxf(h, 0.f);
    p0 += h * w2[j];           p1 += h * w2[512 + j];
    p2 += h * w2[1024 + j];    p3 += h * w2[1536 + j];
  }
  __shared__ float red[4][128];
  red[0][t] = p0; red[1][t] = p1; red[2][t] = p2; red[3][t] = p3;
  __syncthreads();
  if (t < 4) {
    float s = 0.f;
    for (int i = 0; i < 128; ++i) s += red[t][i];
    bt[e * 4 + t] = s;
  }
}

// ---------------------------------------------------------------------------
// K2: QKV projection GEMM  [65536,128] x [384,128]^T  (bf16 WMMA, f32 acc)
// Epilogue scatters into q_ws/k_ws as [win][head][n][d] and v_ws transposed
// as [win][head][d][n] so the PV-stage B fragments are contiguous in LDS.
// ---------------------------------------------------------------------------
__global__ void qkv_gemm_kernel(const __bf16* __restrict__ X,   // [65536][128]
                                const __bf16* __restrict__ W,   // [384][128]
                                const float*  __restrict__ bq,  // [384]
                                __bf16* __restrict__ qw,
                                __bf16* __restrict__ kw,
                                __bf16* __restrict__ vw) {
  __shared__ __bf16 lA[64 * 128];
  __shared__ __bf16 lB[64 * 128];
  const int m0 = blockIdx.x * 64;
  const int n0 = blockIdx.y * 64;
  const int tid = threadIdx.x;

  // async-stage both 16KB tiles (1024 x 16B each, 4 per thread)
  {
    const unsigned aoff = (unsigned)(size_t)lA;
    const unsigned boff = (unsigned)(size_t)lB;
    const void* gA = (const void*)(X + m0 * DIM);
    const void* gB = (const void*)(W + n0 * DIM);
    for (int i = tid; i < 1024; i += 256) {
      async_copy_b128(aoff + i * 16, gA, i * 16);
      async_copy_b128(boff + i * 16, gB, i * 16);
    }
    wait_async0();
  }
  __syncthreads();

  const int lane = tid & 31;
  const int wv   = tid >> 5;
  const int nloc = lane & 15;
  const int mb   = (lane & 16) ? 8 : 0;

#pragma unroll
  for (int tt = 0; tt < 2; ++tt) {
    const int t  = wv + tt * 8;
    const int mt = t & 3, nt = t >> 2;
    v8f acc = {};
#pragma unroll
    for (int kk = 0; kk < 4; ++kk) {
      v16bf a = load_a_frag(lA + (mt * 16) * 128 + kk * 32, 128, lane);
      v16bf b = load_b_frag(lB + (nt * 16) * 128 + kk * 32, 128, lane);
      acc = wmma_bf16(a, b, acc);
    }
    const int c    = n0 + nt * 16 + nloc;      // output column in [0,384)
    const float bc = bq[c];
    const int cc = c & 127, hh = cc >> 5, dd = cc & 31;
#pragma unroll
    for (int r = 0; r < 8; ++r) {
      const int m = m0 + mt * 16 + mb + r;
      const float v = acc[r] + bc;
      const int bwin = m >> 9, nn = m & 511;
      const int bh = bwin * HEADS + hh;
      if (c < 128)       qw[(bh * NTOK + nn) * HDIM + dd] = (__bf16)v;
      else if (c < 256)  kw[(bh * NTOK + nn) * HDIM + dd] = (__bf16)v;
      else               vw[(bh * HDIM + dd) * NTOK + nn] = (__bf16)v;
    }
  }
}

// ---------------------------------------------------------------------------
// K3: attention per (window, head, 64-query tile). 260 KB dynamic LDS.
// ---------------------------------------------------------------------------
__global__ void attn_kernel(const __bf16* __restrict__ qw,
                            const __bf16* __restrict__ kw,
                            const __bf16* __restrict__ vw,
                            const float*  __restrict__ bt,        // [3375][4]
                            const int*    __restrict__ rel_index, // [512*512]
                            const float*  __restrict__ logit_scale,
                            __bf16* __restrict__ ctx) {           // [65536][128]
  extern __shared__ char smem[];
  __bf16* qs = (__bf16*)smem;                    // [64][32]
  __bf16* ks = qs + QTILE * HDIM;                // [512][32]
  __bf16* vs = ks + NTOK * HDIM;                 // [32][512] (V^T)
  float*  S  = (float*)(vs + HDIM * NTOK);       // [64][512]
  __bf16* P  = (__bf16*)(S + QTILE * NTOK);      // [64][512]

  const int qt = blockIdx.x, h = blockIdx.y, bwin = blockIdx.z;
  const int q0 = qt * QTILE;
  const int bh = bwin * HEADS + h;
  const int tid = threadIdx.x;
  const float scale = __expf(fminf(logit_scale[h], 4.6051702f)); // log(100)

  // ---- async-stage Q tile (4KB), K (32KB), V^T (32KB) into LDS ----
  {
    const void* gq = (const void*)(qw + (bh * NTOK + q0) * HDIM);
    const void* gk = (const void*)(kw + bh * NTOK * HDIM);
    const void* gv = (const void*)(vw + bh * HDIM * NTOK);
    const unsigned qoff = (unsigned)(size_t)qs;
    const unsigned koff = (unsigned)(size_t)ks;
    const unsigned voff = (unsigned)(size_t)vs;
    async_copy_b128(qoff + tid * 16, gq, tid * 16);          // 256 x 16B
    for (int i = tid; i < 2048; i += 256) {                  // 2048 x 16B each
      async_copy_b128(koff + i * 16, gk, i * 16);
      async_copy_b128(voff + i * 16, gv, i * 16);
    }
    wait_async0();
  }
  __syncthreads();

  // ---- L2-normalize Q rows (64) and K rows (512) in f32 ----
  if (tid < QTILE) {
    float ss = 0.f;
    for (int d = 0; d < HDIM; ++d) { float x = (float)qs[tid * HDIM + d]; ss += x * x; }
    float inv = 1.f / fmaxf(sqrtf(ss), 1e-12f);
    for (int d = 0; d < HDIM; ++d)
      qs[tid * HDIM + d] = (__bf16)((float)qs[tid * HDIM + d] * inv);
  }
  for (int r = tid; r < NTOK; r += 256) {
    float ss = 0.f;
    for (int d = 0; d < HDIM; ++d) { float x = (float)ks[r * HDIM + d]; ss += x * x; }
    float inv = 1.f / fmaxf(sqrtf(ss), 1e-12f);
    for (int d = 0; d < HDIM; ++d)
      ks[r * HDIM + d] = (__bf16)((float)ks[r * HDIM + d] * inv);
  }
  __syncthreads();

  const int lane = tid & 31;
  const int wv   = tid >> 5;
  const int nloc = lane & 15;
  const int mb   = (lane & 16) ? 8 : 0;

  // ---- S = scale * Qn @ Kn^T : 4x32 tiles of 16x16, single wmma (K=32=d) ----
  {
    const int mt  = wv & 3;
    const int ntb = wv >> 2;
    v16bf a = load_a_frag(qs + (mt * 16) * HDIM, HDIM, lane);
#pragma unroll
    for (int i = 0; i < 16; ++i) {
      const int nt = ntb + (i << 1);
      v16bf b = load_b_frag(ks + (nt * 16) * HDIM, HDIM, lane);
      v8f acc = {};
      acc = wmma_bf16(a, b, acc);
      float* dst = S + (mt * 16) * NTOK + nt * 16 + nloc;
#pragma unroll
      for (int r = 0; r < 8; ++r) dst[(mb + r) * NTOK] = acc[r] * scale;
    }
  }
  __syncthreads();

  // ---- softmax over 512 keys with CPB bias; 4 threads per row ----
  {
    const int row  = tid >> 2;
    const int part = tid & 3;
    float* srow = S + row * NTOK + part * 128;
    const int* ridx = rel_index + (q0 + row) * NTOK + part * 128;
    float m = -1e30f;
    for (int j = 0; j < 128; ++j) {
      float bb = bt[ridx[j] * HEADS + h];
      bb = 16.f / (1.f + __expf(-bb));
      float x = srow[j] + bb;
      srow[j] = x;
      m = fmaxf(m, x);
    }
    m = fmaxf(m, __shfl_xor(m, 1));
    m = fmaxf(m, __shfl_xor(m, 2));
    float s = 0.f;
    for (int j = 0; j < 128; ++j) {
      float e = __expf(srow[j] - m);
      srow[j] = e;
      s += e;
    }
    s += __shfl_xor(s, 1);
    s += __shfl_xor(s, 2);
    const float inv = 1.f / s;
    __bf16* prow = P + row * NTOK + part * 128;
    for (int j = 0; j < 128; ++j) prow[j] = (__bf16)(srow[j] * inv);
  }
  __syncthreads();

  // ---- ctx = P @ V : 8 tiles (4 M x 2 N), 16-step K loop of WMMAs ----
  {
    const int mt = wv & 3;
    const int nt = wv >> 2;      // d-tile (0 or 1)
    v8f acc = {};
#pragma unroll
    for (int kt = 0; kt < 16; ++kt) {
      v16bf a = load_a_frag(P + (mt * 16) * NTOK + kt * 32, NTOK, lane);
      v16bf b = load_b_frag(vs + (nt * 16) * NTOK + kt * 32, NTOK, lane);
      acc = wmma_bf16(a, b, acc);
    }
    const int col = h * HDIM + nt * 16 + nloc;
#pragma unroll
    for (int r = 0; r < 8; ++r) {
      const int row = q0 + mt * 16 + mb + r;
      ctx[(bwin * NTOK + row) * DIM + col] = (__bf16)acc[r];
    }
  }
}

// ---------------------------------------------------------------------------
// K4: output projection GEMM [65536,128] x [128,128]^T + bias -> f32 out
// ---------------------------------------------------------------------------
__global__ void proj_gemm_kernel(const __bf16* __restrict__ A,   // ctx bf16
                                 const __bf16* __restrict__ W,   // [128][128]
                                 const float*  __restrict__ bp,  // [128]
                                 float* __restrict__ out) {      // [65536][128]
  __shared__ __bf16 lA[64 * 128];
  __shared__ __bf16 lB[64 * 128];
  const int m0 = blockIdx.x * 64;
  const int n0 = blockIdx.y * 64;
  const int tid = threadIdx.x;

  {
    const unsigned aoff = (unsigned)(size_t)lA;
    const unsigned boff = (unsigned)(size_t)lB;
    const void* gA = (const void*)(A + m0 * DIM);
    const void* gB = (const void*)(W + n0 * DIM);
    for (int i = tid; i < 1024; i += 256) {
      async_copy_b128(aoff + i * 16, gA, i * 16);
      async_copy_b128(boff + i * 16, gB, i * 16);
    }
    wait_async0();
  }
  __syncthreads();

  const int lane = tid & 31;
  const int wv   = tid >> 5;
  const int nloc = lane & 15;
  const int mb   = (lane & 16) ? 8 : 0;

#pragma unroll
  for (int tt = 0; tt < 2; ++tt) {
    const int t  = wv + tt * 8;
    const int mt = t & 3, nt = t >> 2;
    v8f acc = {};
#pragma unroll
    for (int kk = 0; kk < 4; ++kk) {
      v16bf a = load_a_frag(lA + (mt * 16) * 128 + kk * 32, 128, lane);
      v16bf b = load_b_frag(lB + (nt * 16) * 128 + kk * 32, 128, lane);
      acc = wmma_bf16(a, b, acc);
    }
    const int c = n0 + nt * 16 + nloc;
    const float bc = bp[c];
#pragma unroll
    for (int r = 0; r < 8; ++r) {
      const int m = m0 + mt * 16 + mb + r;
      out[m * DIM + c] = acc[r] + bc;
    }
  }
}

// ---------------------------------------------------------------------------
extern "C" void kernel_launch(void* const* d_in, const int* in_sizes, int n_in,
                              void* d_out, int out_size, void* d_ws, size_t ws_size,
                              hipStream_t stream) {
  const float* hs      = (const float*)d_in[0];
  const float* Wqkv    = (const float*)d_in[1];
  const float* bqkv    = (const float*)d_in[2];
  const float* Wproj   = (const float*)d_in[3];
  const float* bproj   = (const float*)d_in[4];
  const float* lscale  = (const float*)d_in[5];
  const float* cpb_w1  = (const float*)d_in[6];
  const float* cpb_b1  = (const float*)d_in[7];
  const float* cpb_w2  = (const float*)d_in[8];
  const float* rtable  = (const float*)d_in[9];
  const int*   rindex  = (const int*)d_in[10];
  float* out = (float*)d_out;

  // workspace layout (256B-aligned slices)
  char* ws = (char*)d_ws;
  size_t off = 0;
  auto take = [&](size_t bytes) {
    char* p = ws + off;
    off = (off + bytes + 255) & ~(size_t)255;
    return p;
  };
  float*  bt    = (float*)take((size_t)TBL * HEADS * sizeof(float));
  __bf16* xbf   = (__bf16*)take((size_t)NWIN * NTOK * DIM * 2);
  __bf16* wqb   = (__bf16*)take((size_t)3 * DIM * DIM * 2);
  __bf16* wpb   = (__bf16*)take((size_t)DIM * DIM * 2);
  __bf16* q_ws  = (__bf16*)take((size_t)NWIN * NTOK * DIM * 2);
  __bf16* k_ws  = (__bf16*)take((size_t)NWIN * NTOK * DIM * 2);
  __bf16* v_ws  = (__bf16*)take((size_t)NWIN * NTOK * DIM * 2);
  __bf16* c_ws  = (__bf16*)take((size_t)NWIN * NTOK * DIM * 2);
  if (off > ws_size) return;  // insufficient scratch; bail deterministically

  // K0: bf16 staging
  {
    const int total = NWIN * NTOK * DIM + 3 * DIM * DIM + DIM * DIM;
    prep_kernel<<<(total + 255) / 256, 256, 0, stream>>>(hs, Wqkv, Wproj, xbf, wqb, wpb);
  }
  // K1: CPB bias table
  cpb_kernel<<<TBL, 128, 0, stream>>>(rtable, cpb_w1, cpb_b1, cpb_w2, bt);
  // K2: QKV GEMM
  {
    dim3 g(NWIN * NTOK / 64, (3 * DIM) / 64);
    qkv_gemm_kernel<<<g, 256, 0, stream>>>(xbf, wqb, bqkv, q_ws, k_ws, v_ws);
  }
  // K3: attention (260 KB dynamic LDS, one WG per WGP)
  {
    const int lds_bytes = (QTILE * HDIM + NTOK * HDIM + HDIM * NTOK) * 2 +
                          QTILE * NTOK * 4 + QTILE * NTOK * 2;   // 266240
    (void)hipFuncSetAttribute((const void*)attn_kernel,
                              hipFuncAttributeMaxDynamicSharedMemorySize, lds_bytes);
    dim3 g(NTOK / QTILE, HEADS, NWIN);
    attn_kernel<<<g, 256, lds_bytes, stream>>>(q_ws, k_ws, v_ws, bt, rindex, lscale, c_ws);
  }
  // K4: output projection
  {
    dim3 g(NWIN * NTOK / 64, DIM / 64);
    proj_gemm_kernel<<<g, 256, 0, stream>>>(c_ws, wpb, bproj, out);
  }
}
